// WanSelfAttention_2465311228596
// MI455X (gfx1250) — matrix operand order repaired and checked by reading, not compiled, not run on hardware
//
#include <hip/hip_runtime.h>
#include <hip/hip_bf16.h>

typedef __bf16 bf16_t;
typedef __attribute__((ext_vector_type(16))) __bf16 v16bf;
typedef __attribute__((ext_vector_type(8)))  __bf16 v8bf;
typedef __attribute__((ext_vector_type(8)))  float  v8f;
typedef __attribute__((ext_vector_type(4)))  float  v4f;
typedef __attribute__((ext_vector_type(4)))  unsigned int u32x4;

#define DIM 2048
#define NH  16
#define HD  128

static __device__ __forceinline__ v16bf make16(v8bf lo, v8bf hi) {
  return __builtin_shufflevector(lo, hi, 0,1,2,3,4,5,6,7,8,9,10,11,12,13,14,15);
}

// ---------------------------------------------------------------- casts
__global__ __launch_bounds__(256)
void cast_f32_to_bf16(const float* __restrict__ in, bf16_t* __restrict__ out, int n8) {
  int g = blockIdx.x * blockDim.x + threadIdx.x;
  if (g >= n8) return;
  const v4f* p = (const v4f*)(in + (size_t)g * 8);
  v4f a = p[0], b = p[1];
  v8bf o;
  o[0]=(bf16_t)a[0]; o[1]=(bf16_t)a[1]; o[2]=(bf16_t)a[2]; o[3]=(bf16_t)a[3];
  o[4]=(bf16_t)b[0]; o[5]=(bf16_t)b[1]; o[6]=(bf16_t)b[2]; o[7]=(bf16_t)b[3];
  *(v8bf*)(out + (size_t)g * 8) = o;
}

// ---------------------------------------------------------------- GEMM: C = A[M,K] * B[N,K]^T + bias
// Block tile 128(M) x 256(N), wave tile 64x64, double-buffered LDS.
#define GSTR 40   // padded LDS row stride (bf16 elems); 80B, 16B aligned

__global__ __launch_bounds__(256)
void gemm_bf16_wmma(const bf16_t* __restrict__ A, const bf16_t* __restrict__ B,
                    const float* __restrict__ bias, float* __restrict__ C,
                    int M, int N, int K) {
  __shared__ bf16_t lA[2][128 * GSTR];
  __shared__ bf16_t lB[2][256 * GSTR];
  const int tid  = threadIdx.x;
  const int lane = tid & 31;
  const int wave = tid >> 5;
  const int m0 = blockIdx.y * 128;
  const int n0 = blockIdx.x * 256;
  const int wm = (wave & 1) * 64;     // 2 waves along M
  const int wn = (wave >> 1) * 64;    // 4 waves along N

  v8f vzero;
#pragma unroll
  for (int e = 0; e < 8; e++) vzero[e] = 0.0f;
  v8f acc[4][4];
#pragma unroll
  for (int i = 0; i < 4; i++)
#pragma unroll
    for (int j = 0; j < 4; j++) acc[i][j] = vzero;

  const int lr = tid >> 2;        // 0..63
  const int lc = (tid & 3) * 8;   // 0,8,16,24
  const int frow = lane & 15;
  const int kh   = (lane >> 4) * 8;
  const int bk   = (lane >> 4) * 16;

  const int nk = K / 32;
  u32x4 ra[2], rb[4];

  // prologue: stage k-step 0 into buffer 0
  {
    const bf16_t* Ap = A + (size_t)(m0 + lr) * K + lc;
    ra[0] = *(const u32x4*)(Ap);
    ra[1] = *(const u32x4*)(Ap + (size_t)64 * K);
    const bf16_t* Bp = B + (size_t)(n0 + lr) * K + lc;
    rb[0] = *(const u32x4*)(Bp);
    rb[1] = *(const u32x4*)(Bp + (size_t)64 * K);
    rb[2] = *(const u32x4*)(Bp + (size_t)128 * K);
    rb[3] = *(const u32x4*)(Bp + (size_t)192 * K);
    *(u32x4*)(&lA[0][lr * GSTR + lc])         = ra[0];
    *(u32x4*)(&lA[0][(lr + 64) * GSTR + lc])  = ra[1];
    *(u32x4*)(&lB[0][lr * GSTR + lc])         = rb[0];
    *(u32x4*)(&lB[0][(lr + 64) * GSTR + lc])  = rb[1];
    *(u32x4*)(&lB[0][(lr + 128) * GSTR + lc]) = rb[2];
    *(u32x4*)(&lB[0][(lr + 192) * GSTR + lc]) = rb[3];
  }
  __syncthreads();

  for (int kk = 0; kk < nk; kk++) {
    const int cur = kk & 1;
    const int nxt = cur ^ 1;
    const bool hasnext = (kk + 1) < nk;
    if (hasnext) {
      const int k0 = (kk + 1) * 32;
      const bf16_t* Ap = A + (size_t)(m0 + lr) * K + k0 + lc;
      ra[0] = *(const u32x4*)(Ap);
      ra[1] = *(const u32x4*)(Ap + (size_t)64 * K);
      const bf16_t* Bp = B + (size_t)(n0 + lr) * K + k0 + lc;
      rb[0] = *(const u32x4*)(Bp);
      rb[1] = *(const u32x4*)(Bp + (size_t)64 * K);
      rb[2] = *(const u32x4*)(Bp + (size_t)128 * K);
      rb[3] = *(const u32x4*)(Bp + (size_t)192 * K);
    }

    v16bf afrag[4], bfrag[4];
#pragma unroll
    for (int i = 0; i < 4; i++) {
      const bf16_t* p = &lA[cur][(wm + i * 16 + frow) * GSTR];
      afrag[i] = make16(*(const v8bf*)(p + kh), *(const v8bf*)(p + kh + 16));
    }
#pragma unroll
    for (int j = 0; j < 4; j++) {
      const bf16_t* p = &lB[cur][(wn + j * 16 + frow) * GSTR + kh];
      bfrag[j] = make16(*(const v8bf*)(p), *(const v8bf*)(p + 16));
    }
#pragma unroll
    for (int i = 0; i < 4; i++)
#pragma unroll
      for (int j = 0; j < 4; j++)
        acc[i][j] = __builtin_amdgcn_wmma_f32_16x16x32_bf16(
            false, afrag[i], false, bfrag[j], (short)0, acc[i][j], false, false);

    if (hasnext) {
      *(u32x4*)(&lA[nxt][lr * GSTR + lc])         = ra[0];
      *(u32x4*)(&lA[nxt][(lr + 64) * GSTR + lc])  = ra[1];
      *(u32x4*)(&lB[nxt][lr * GSTR + lc])         = rb[0];
      *(u32x4*)(&lB[nxt][(lr + 64) * GSTR + lc])  = rb[1];
      *(u32x4*)(&lB[nxt][(lr + 128) * GSTR + lc]) = rb[2];
      *(u32x4*)(&lB[nxt][(lr + 192) * GSTR + lc]) = rb[3];
    }
    __syncthreads();
  }

  const int crow = (lane >> 4) * 8;
#pragma unroll
  for (int i = 0; i < 4; i++)
#pragma unroll
    for (int j = 0; j < 4; j++) {
      int col = n0 + wn + j * 16 + frow;
      float bv = bias[col];
#pragma unroll
      for (int e = 0; e < 8; e++) {
        int row = m0 + wm + i * 16 + crow + e;
        C[(size_t)row * N + col] = acc[i][j][e] + bv;
      }
    }
}

// ---------------------------------------------------------------- RMSNorm + 3D RoPE + pack to [B,NH,L,HD] bf16
__global__ __launch_bounds__(256)
void rmsnorm_rope_pack(const float* __restrict__ X, const float* __restrict__ g,
                       const float* __restrict__ fcos, const float* __restrict__ fsin,
                       const int* __restrict__ pF, const int* __restrict__ pH,
                       const int* __restrict__ pW,
                       bf16_t* __restrict__ out, int Lctx) {
  const int token = blockIdx.x;
  const int b = token / Lctx;
  const int l = token - b * Lctx;
  const int tid = threadIdx.x;
  const float* xr = X + (size_t)token * DIM;
  const int cb = tid * 8;

  float v[8];
  v4f a = *(const v4f*)(xr + cb), bb = *(const v4f*)(xr + cb + 4);
  v[0]=a[0]; v[1]=a[1]; v[2]=a[2]; v[3]=a[3];
  v[4]=bb[0]; v[5]=bb[1]; v[6]=bb[2]; v[7]=bb[3];

  float ss = 0.0f;
#pragma unroll
  for (int i = 0; i < 8; i++) ss += v[i] * v[i];
  __shared__ float red[256];
  red[tid] = ss;
  __syncthreads();
  for (int s = 128; s > 0; s >>= 1) {
    if (tid < s) red[tid] += red[tid + s];
    __syncthreads();
  }
  float scale = rsqrtf(red[0] * (1.0f / DIM) + 1e-6f);
#pragma unroll
  for (int i = 0; i < 8; i++) v[i] = v[i] * scale * g[cb + i];

  const int F = *pF, H = *pH, W = *pW;
  (void)F;
  const int f  = l / (H * W);
  const int hh = (l / W) % H;
  const int ww = l % W;
  const int c  = HD / 2;          // 64
  const int c2 = c / 3;           // 21
  const int c1 = c - 2 * c2;      // 22
  const int hidx = cb >> 7;
  const int j = cb & 127;
  const int p0 = j >> 1;
#pragma unroll
  for (int pi = 0; pi < 4; pi++) {
    int p = p0 + pi;
    int row = (p < c1) ? f : ((p < c1 + c2) ? hh : ww);
    float fc = fcos[row * c + p];
    float fs = fsin[row * c + p];
    float xr_ = v[2 * pi], xi_ = v[2 * pi + 1];
    v[2 * pi]     = xr_ * fc - xi_ * fs;
    v[2 * pi + 1] = xr_ * fs + xi_ * fc;
  }
  v8bf o;
#pragma unroll
  for (int i = 0; i < 8; i++) o[i] = (bf16_t)v[i];
  *(v8bf*)(out + (((size_t)(b * NH + hidx) * Lctx) + l) * HD + j) = o;
}

// ---------------------------------------------------------------- V repack: [B*L,DIM] f32 -> [B,NH,L,HD] bf16
__global__ __launch_bounds__(256)
void vcast_pack(const float* __restrict__ X, bf16_t* __restrict__ out, int Lctx) {
  int gt = blockIdx.x * blockDim.x + threadIdx.x;   // one per 8 elems
  int token = gt >> 8;            // DIM/8 = 256 chunks per token
  int cb = (gt & 255) * 8;
  int b = token / Lctx;
  int l = token - b * Lctx;
  int hidx = cb >> 7;
  int j = cb & 127;
  const float* xr = X + (size_t)token * DIM + cb;
  v4f a = *(const v4f*)(xr), bb = *(const v4f*)(xr + 4);
  v8bf o;
  o[0]=(bf16_t)a[0]; o[1]=(bf16_t)a[1]; o[2]=(bf16_t)a[2]; o[3]=(bf16_t)a[3];
  o[4]=(bf16_t)bb[0]; o[5]=(bf16_t)bb[1]; o[6]=(bf16_t)bb[2]; o[7]=(bf16_t)bb[3];
  *(v8bf*)(out + (((size_t)(b * NH + hidx) * Lctx) + l) * HD + j) = o;
}

// ---------------------------------------------------------------- Flash attention (key block = 64)
#define KSTR 136   // K-tile LDS row stride (bf16)
#define VSTR 72    // V^T / P LDS row stride (bf16); 144B, 16B aligned

__global__ __launch_bounds__(256)
void flash_attn(const bf16_t* __restrict__ Q, const bf16_t* __restrict__ Kd,
                const bf16_t* __restrict__ V, const int* __restrict__ seq_lens,
                bf16_t* __restrict__ O, int Lctx) {
  __shared__ bf16_t lK[64 * KSTR];       // [key][d]
  __shared__ bf16_t lVT[HD * VSTR];      // [d][key]
  __shared__ bf16_t lP[8 * 16 * VSTR];   // per-wave [16][keys]

  const int tid  = threadIdx.x;
  const int lane = tid & 31;
  const int wave = tid >> 5;
  const int bh = blockIdx.y;
  const int b  = bh >> 4;
  const int h  = bh & 15;
  const int q0 = blockIdx.x * 128 + wave * 16;
  const int slen = seq_lens[b];

  const bf16_t* qbase = Q  + (size_t)bh * Lctx * HD;
  const bf16_t* kbase = Kd + (size_t)bh * Lctx * HD;
  const bf16_t* vbase = V  + (size_t)bh * Lctx * HD;

  const int frow = lane & 15;
  const int kh   = (lane >> 4) * 8;
  const int bk   = (lane >> 4) * 16;
  const int crow = (lane >> 4) * 8;

  // Q fragments: 16 rows x full d=128 (4 chunks of K=32), kept in registers
  v16bf qf[4];
#pragma unroll
  for (int cch = 0; cch < 4; cch++) {
    const bf16_t* p = qbase + (size_t)(q0 + frow) * HD + cch * 32;
    qf[cch] = make16(*(const v8bf*)(p + kh), *(const v8bf*)(p + kh + 16));
  }

  v8f vzero;
#pragma unroll
  for (int e = 0; e < 8; e++) vzero[e] = 0.0f;
  v8f oacc[8];
#pragma unroll
  for (int t = 0; t < 8; t++) oacc[t] = vzero;
  float m_r[8], l_r[8];
#pragma unroll
  for (int e = 0; e < 8; e++) { m_r[e] = -__builtin_inff(); l_r[e] = 0.0f; }

  const float scale = 0.08838834764831845f;   // 1/sqrt(128)
  const int sr = tid >> 2;            // key row 0..63
  const int sc = (tid & 3) * 32;      // d chunk base

  for (int kb = 0; kb < Lctx; kb += 64) {
    // stage K block [64][128] row-major
    {
      const bf16_t* kp = kbase + (size_t)(kb + sr) * HD + sc;
      *(v8bf*)(lK + sr * KSTR + sc)      = *(const v8bf*)(kp);
      *(v8bf*)(lK + sr * KSTR + sc + 8)  = *(const v8bf*)(kp + 8);
      *(v8bf*)(lK + sr * KSTR + sc + 16) = *(const v8bf*)(kp + 16);
      *(v8bf*)(lK + sr * KSTR + sc + 24) = *(const v8bf*)(kp + 24);
      // stage V transposed [128][64]
      const bf16_t* vp = vbase + (size_t)(kb + sr) * HD + sc;
      v8bf va0 = *(const v8bf*)(vp);
      v8bf va1 = *(const v8bf*)(vp + 8);
      v8bf va2 = *(const v8bf*)(vp + 16);
      v8bf va3 = *(const v8bf*)(vp + 24);
#pragma unroll
      for (int jj = 0; jj < 8; jj++) {
        lVT[(sc + jj) * VSTR + sr]      = va0[jj];
        lVT[(sc + 8 + jj) * VSTR + sr]  = va1[jj];
        lVT[(sc + 16 + jj) * VSTR + sr] = va2[jj];
        lVT[(sc + 24 + jj) * VSTR + sr] = va3[jj];
      }
    }
    __syncthreads();

    // S = Q * K^T (16 x 64) as four 16x16 tiles
    v8f s[4];
#pragma unroll
    for (int n16 = 0; n16 < 4; n16++) s[n16] = vzero;
#pragma unroll
    for (int n16 = 0; n16 < 4; n16++) {
#pragma unroll
      for (int cch = 0; cch < 4; cch++) {
        const bf16_t* p = lK + (n16 * 16 + frow) * KSTR + cch * 32 + kh;
        v16bf kfrag = make16(*(const v8bf*)(p), *(const v8bf*)(p + 16));
        s[n16] = __builtin_amdgcn_wmma_f32_16x16x32_bf16(
            false, qf[cch], false, kfrag, (short)0, s[n16], false, false);
      }
    }
    // scale + mask
#pragma unroll
    for (int n16 = 0; n16 < 4; n16++) {
      int keyg = kb + n16 * 16 + frow;
      bool valid = keyg < slen;
#pragma unroll
      for (int e = 0; e < 8; e++) {
        float sv = s[n16][e] * scale;
        s[n16][e] = valid ? sv : -__builtin_inff();
      }
    }
    // online softmax
    float alpha[8];
#pragma unroll
    for (int e = 0; e < 8; e++) {
      float v = fmaxf(fmaxf(s[0][e], s[1][e]), fmaxf(s[2][e], s[3][e]));
      v = fmaxf(v, __shfl_xor(v, 1));
      v = fmaxf(v, __shfl_xor(v, 2));
      v = fmaxf(v, __shfl_xor(v, 4));
      v = fmaxf(v, __shfl_xor(v, 8));
      float mn = fmaxf(m_r[e], v);
      alpha[e] = __expf(m_r[e] - mn);
      m_r[e] = mn;
    }
#pragma unroll
    for (int e = 0; e < 8; e++) {
      float p0 = __expf(s[0][e] - m_r[e]);
      float p1 = __expf(s[1][e] - m_r[e]);
      float p2 = __expf(s[2][e] - m_r[e]);
      float p3 = __expf(s[3][e] - m_r[e]);
      s[0][e] = p0; s[1][e] = p1; s[2][e] = p2; s[3][e] = p3;
      float rs = (p0 + p1) + (p2 + p3);
      rs += __shfl_xor(rs, 1);
      rs += __shfl_xor(rs, 2);
      rs += __shfl_xor(rs, 4);
      rs += __shfl_xor(rs, 8);
      l_r[e] = l_r[e] * alpha[e] + rs;
    }
#pragma unroll
    for (int t = 0; t < 8; t++)
#pragma unroll
      for (int e = 0; e < 8; e++) oacc[t][e] *= alpha[e];

    // P (16x64 bf16) -> per-wave LDS, transpose C-layout into A-layout
    bf16_t* pw = lP + wave * 16 * VSTR;
#pragma unroll
    for (int n16 = 0; n16 < 4; n16++)
#pragma unroll
      for (int e = 0; e < 8; e++)
        pw[(crow + e) * VSTR + n16 * 16 + frow] = (bf16_t)s[n16][e];
    asm volatile("s_wait_dscnt 0" ::: "memory");

    v16bf pfrag[2];
#pragma unroll
    for (int cch = 0; cch < 2; cch++) {
      const bf16_t* pr = pw + frow * VSTR + cch * 32 + kh;
      pfrag[cch] = make16(*(const v8bf*)(pr), *(const v8bf*)(pr + 16));
    }

    // O += P * V  (8 output tiles of 16 d-columns, K=64 over 2 chunks)
#pragma unroll
    for (int t = 0; t < 8; t++) {
#pragma unroll
      for (int cch = 0; cch < 2; cch++) {
        const bf16_t* vpp = lVT + (t * 16 + frow) * VSTR + cch * 32 + kh;
        v16bf vfrag = make16(*(const v8bf*)(vpp), *(const v8bf*)(vpp + 16));
        oacc[t] = __builtin_amdgcn_wmma_f32_16x16x32_bf16(
            false, pfrag[cch], false, vfrag, (short)0, oacc[t], false, false);
      }
    }
    __syncthreads();
  }

  // epilogue: O /= l ; store to [B, L, NH*HD] bf16
  float inv[8];
#pragma unroll
  for (int e = 0; e < 8; e++) inv[e] = (l_r[e] > 0.0f) ? 1.0f / l_r[e] : 0.0f;
#pragma unroll
  for (int t = 0; t < 8; t++)
#pragma unroll
    for (int e = 0; e < 8; e++) {
      int qrow = q0 + crow + e;
      int col = h * HD + t * 16 + frow;
      O[((size_t)(b * Lctx + qrow)) * DIM + col] = (bf16_t)(oacc[t][e] * inv[e]);
    }
}

// ---------------------------------------------------------------- host launcher
extern "C" void kernel_launch(void* const* d_in, const int* in_sizes, int n_in,
                              void* d_out, int out_size, void* d_ws, size_t ws_size,
                              hipStream_t stream) {
  (void)n_in; (void)out_size; (void)ws_size;
  const float* x   = (const float*)d_in[0];
  const float* Wq  = (const float*)d_in[1];
  const float* bq  = (const float*)d_in[2];
  const float* Wk  = (const float*)d_in[3];
  const float* bk  = (const float*)d_in[4];
  const float* Wv  = (const float*)d_in[5];
  const float* bv  = (const float*)d_in[6];
  const float* Wo  = (const float*)d_in[7];
  const float* bo  = (const float*)d_in[8];
  const float* gq  = (const float*)d_in[9];
  const float* gk  = (const float*)d_in[10];
  const int*   seq = (const int*)d_in[11];
  const int*   pF  = (const int*)d_in[12];
  const int*   pH  = (const int*)d_in[13];
  const int*   pW  = (const int*)d_in[14];
  const float* fc  = (const float*)d_in[15];
  const float* fs  = (const float*)d_in[16];

  const int B = in_sizes[11];
  const int M = in_sizes[0] / DIM;   // B*L
  const int L = M / B;
  const int K = DIM, N = DIM;

  char* ws = (char*)d_ws;
  size_t off = 0;
  auto take = [&](size_t bytes) -> char* {
    char* p = ws + off;
    off += (bytes + 255) & ~(size_t)255;
    return p;
  };
  bf16_t* xb  = (bf16_t*)take((size_t)M * K * 2);
  bf16_t* wqb = (bf16_t*)take((size_t)N * K * 2);
  bf16_t* wkb = (bf16_t*)take((size_t)N * K * 2);
  bf16_t* wvb = (bf16_t*)take((size_t)N * K * 2);
  bf16_t* wob = (bf16_t*)take((size_t)N * K * 2);
  float*  tf  = (float*)take((size_t)M * N * 4);
  bf16_t* qb  = (bf16_t*)take((size_t)M * DIM * 2);
  bf16_t* kbf = (bf16_t*)take((size_t)M * DIM * 2);
  bf16_t* vbf = (bf16_t*)take((size_t)M * DIM * 2);
  bf16_t* ob  = (bf16_t*)take((size_t)M * DIM * 2);

  dim3 blk(256);
  int nx8 = M * K / 8;
  cast_f32_to_bf16<<<nx8 / 256, blk, 0, stream>>>(x, xb, nx8);
  int nw8 = N * K / 8;
  cast_f32_to_bf16<<<nw8 / 256, blk, 0, stream>>>(Wq, wqb, nw8);
  cast_f32_to_bf16<<<nw8 / 256, blk, 0, stream>>>(Wk, wkb, nw8);
  cast_f32_to_bf16<<<nw8 / 256, blk, 0, stream>>>(Wv, wvb, nw8);
  cast_f32_to_bf16<<<nw8 / 256, blk, 0, stream>>>(Wo, wob, nw8);

  dim3 ggrid(N / 256, M / 128);
  // Q
  gemm_bf16_wmma<<<ggrid, blk, 0, stream>>>(xb, wqb, bq, tf, M, N, K);
  rmsnorm_rope_pack<<<M, blk, 0, stream>>>(tf, gq, fc, fs, pF, pH, pW, qb, L);
  // K
  gemm_bf16_wmma<<<ggrid, blk, 0, stream>>>(xb, wkb, bk, tf, M, N, K);
  rmsnorm_rope_pack<<<M, blk, 0, stream>>>(tf, gk, fc, fs, pF, pH, pW, kbf, L);
  // V
  gemm_bf16_wmma<<<ggrid, blk, 0, stream>>>(xb, wvb, bv, tf, M, N, K);
  vcast_pack<<<(M * DIM / 8) / 256, blk, 0, stream>>>(tf, vbf, L);
  // attention
  dim3 agrid(L / 128, B * NH);
  flash_attn<<<agrid, blk, 0, stream>>>(qb, kbf, vbf, seq, ob, L);
  // output projection -> fp32 out
  gemm_bf16_wmma<<<ggrid, blk, 0, stream>>>(ob, wob, bo, (float*)d_out, M, N, K);
}
